// ScaledDotProductAttention_56032143343618
// MI455X (gfx1250) — matrix operand order repaired and checked by reading, not compiled, not run on hardware
//
#include <hip/hip_runtime.h>
#include <hip/hip_bf16.h>

// ============================================================================
// ScaledDotProductAttention scores+softmax for MI455X (gfx1250, wave32, WMMA)
//
// out = softmax( (x@w_q) @ (x@w_k)^T / sqrt(1024) )   [4096 x 4096] f32
//
// Compute-bound on the 4096x4096x1024 scores GEMM -> bf16 WMMA
// (v_wmma_f32_16x16x32_bf16) with bf16x3 split accumulation (hi*hi + hi*lo +
// lo*hi) for ~fp32-class accuracy at 3/8 the cost of native f32 WMMA.
//
// Double-buffered LDS staging via GLOBAL_LOAD_ASYNC_TO_LDS_B128 (ASYNCcnt),
// global_prefetch of the k+2 tile. One barrier per k-step; tile i+1 staged
// while tile i runs through the WMMAs.
// ============================================================================

#define SEQ_LEN 4096
#define DMODEL  1024

typedef __attribute__((ext_vector_type(16))) __bf16 v16bf;
typedef __attribute__((ext_vector_type(8)))  float  v8f;
typedef __attribute__((ext_vector_type(4)))  int    v4i;

struct U16x16 { unsigned short v[16]; };

#if defined(__gfx1250__) && __has_builtin(__builtin_amdgcn_global_load_async_to_lds_b128) && __has_builtin(__builtin_amdgcn_s_wait_asynccnt)
#define USE_ASYNC_LDS 1
#else
#define USE_ASYNC_LDS 0
#endif

__device__ __forceinline__ unsigned short f32_to_bf16(float f) {
    unsigned int u = __float_as_uint(f);
    u += 0x7FFFu + ((u >> 16) & 1u);          // round-to-nearest-even
    return (unsigned short)(u >> 16);
}
__device__ __forceinline__ float bf16_to_f32(unsigned short h) {
    return __uint_as_float(((unsigned int)h) << 16);
}

// 16-byte global -> LDS copy: async (no VGPR round trip, ASYNCcnt) if
// available, else synchronous through registers.
__device__ __forceinline__ void copy16_g2l(const unsigned short* g, unsigned short* l) {
#if USE_ASYNC_LDS
    // Builtin signature (from hipcc diagnostic): params are v4i* (generic).
    __builtin_amdgcn_global_load_async_to_lds_b128((v4i*)g, (v4i*)l, 0, 0);
#else
    *(uint4*)l = *(const uint4*)g;
#endif
}
__device__ __forceinline__ void wait_g2l() {
#if USE_ASYNC_LDS
    __builtin_amdgcn_s_wait_asynccnt(0);
#endif
}

// ---------------------------------------------------------------------------
// Kernel 0: split f32 array into bf16 hi/lo pair (grid-stride)
// ---------------------------------------------------------------------------
__global__ __launch_bounds__(256)
void split_bf16_kernel(const float* __restrict__ in,
                       unsigned short* __restrict__ hi,
                       unsigned short* __restrict__ lo, int n) {
    int i = blockIdx.x * blockDim.x + threadIdx.x;
    int stride = gridDim.x * blockDim.x;
    for (; i < n; i += stride) {
        float v = in[i];
        unsigned short h = f32_to_bf16(v);
        hi[i] = h;
        lo[i] = f32_to_bf16(v - bf16_to_f32(h));
    }
}

// ---------------------------------------------------------------------------
// bf16x3 split GEMM, double-buffered async staging.
//   Workgroup: 256 threads = 8 waves -> 128(M) x 64(N) tile.
//   Wave: 32x32 = 2x2 WMMA 16x16 tiles, K-loop step 32.
//   TRANSB:    B[k][n] = Bsrc[n*ldb + k]  (scores path; transpose done by the
//              scalar staging path since async copies cannot transpose)
//   OUT_SPLIT: write bf16 hi/lo split of result (projection path)
// LDS: A 2buf x hi/lo x 128x32  = 32 KB, B 2buf x hi/lo x 32x72 = 18 KB.
// ---------------------------------------------------------------------------
template <bool TRANSB, bool OUT_SPLIT>
__global__ __launch_bounds__(256)
void gemm_bf16x3_kernel(const unsigned short* __restrict__ Ahi,
                        const unsigned short* __restrict__ Alo,
                        const unsigned short* __restrict__ Bhi,
                        const unsigned short* __restrict__ Blo,
                        int Kdim, int lda, int ldb, int ldo, float scale,
                        float* __restrict__ outF,
                        unsigned short* __restrict__ outHi,
                        unsigned short* __restrict__ outLo) {
    __shared__ __attribute__((aligned(16))) unsigned short AsHi[2][128 * 32];
    __shared__ __attribute__((aligned(16))) unsigned short AsLo[2][128 * 32];
    __shared__ __attribute__((aligned(16))) unsigned short BsHi[2][32 * 72]; // [k][n], +8 pad
    __shared__ __attribute__((aligned(16))) unsigned short BsLo[2][32 * 72];

    const int tid   = threadIdx.x;
    const int lane  = tid & 31;
    const int wid   = tid >> 5;
    const int waveM = wid >> 1;           // 0..3
    const int waveN = wid & 1;            // 0..1
    const int Mbase = blockIdx.x * 128;
    const int Nbase = blockIdx.y * 64;

    const int laneM = lane & 15;          // A-frag: row within 16x16
    const int khalf = lane >> 4;          // A-frag: K half selector
    const int bRow  = lane & 15;          // B-frag: N column 0..15

    // Stage one 128x32 A tile (hi+lo) and one 32x64 B tile (hi+lo) into buf.
    auto stage = [&](int buf, int kb) {
        // A: 512 chunks of 8 bf16 per half; 2 chunks per thread per half.
#pragma unroll
        for (int c = 0; c < 2; ++c) {
            const int cc  = tid * 2 + c;
            const int row = cc >> 2;
            const int col = (cc & 3) * 8;
            const size_t g = (size_t)(Mbase + row) * lda + kb + col;
            copy16_g2l(Ahi + g, &AsHi[buf][row * 32 + col]);
            copy16_g2l(Alo + g, &AsLo[buf][row * 32 + col]);
            if (c == 0 && kb + 32 < Kdim)   // pull k+2 tile toward L2
                __builtin_prefetch((const void*)(Ahi + g + 32), 0, 1);
        }
        if (TRANSB) {
            // Bs[k][n] = Bsrc[(Nbase+n)*ldb + kb + k] : coalesced read along k,
            // scalar transpose into LDS (async path cannot transpose).
            const int n  = tid >> 2;          // 0..63
            const int kg = (tid & 3) * 8;     // 0,8,16,24
            const size_t g = (size_t)(Nbase + n) * ldb + kb + kg;
            if (kb + 32 < Kdim)
                __builtin_prefetch((const void*)(Bhi + g + 32), 0, 1);
#pragma unroll
            for (int j = 0; j < 8; ++j) {
                BsHi[buf][(kg + j) * 72 + n] = Bhi[g + j];
                BsLo[buf][(kg + j) * 72 + n] = Blo[g + j];
            }
        } else {
            // Bs[k][n] = Bsrc[(kb+k)*ldb + Nbase + n] : contiguous, async.
            const int k  = tid >> 3;          // 0..31
            const int ng = (tid & 7) * 8;     // 0..56
            const size_t g = (size_t)(kb + k) * ldb + Nbase + ng;
            copy16_g2l(Bhi + g, &BsHi[buf][k * 72 + ng]);
            copy16_g2l(Blo + g, &BsLo[buf][k * 72 + ng]);
        }
    };

    v8f acc[2][2];
#pragma unroll
    for (int mt = 0; mt < 2; ++mt)
#pragma unroll
        for (int nt = 0; nt < 2; ++nt)
#pragma unroll
            for (int r = 0; r < 8; ++r) acc[mt][nt][r] = 0.0f;

    stage(0, 0);
    int cur = 0;
    for (int kb = 0; kb < Kdim; kb += 32) {
        wait_g2l();          // this wave's async copies for tile `cur` done
        __syncthreads();     // all waves' copies done; prev compute retired

        if (kb + 32 < Kdim) stage(cur ^ 1, kb + 32);   // overlap with compute

        // ---- A fragments (16x32 bf16 layout per ISA 7.12.2)
        v16bf aHi[2], aLo[2];
#pragma unroll
        for (int mt = 0; mt < 2; ++mt) {
            const int row = (waveM * 32 + mt * 16 + laneM) * 32;
            U16x16 th, tl;
#pragma unroll
            for (int e = 0; e < 8; ++e) {          // K = 8*khalf + e
                th.v[e] = AsHi[cur][row + 8 * khalf + e];
                tl.v[e] = AsLo[cur][row + 8 * khalf + e];
            }
#pragma unroll
            for (int e = 0; e < 8; ++e) {          // K = 16 + 8*khalf + e
                th.v[8 + e] = AsHi[cur][row + 16 + 8 * khalf + e];
                tl.v[8 + e] = AsLo[cur][row + 16 + 8 * khalf + e];
            }
            aHi[mt] = __builtin_bit_cast(v16bf, th);
            aLo[mt] = __builtin_bit_cast(v16bf, tl);
        }
        // ---- B fragments (32x16 bf16: lane -> N column, element -> K, same order as A)
        v16bf bHi[2], bLo[2];
#pragma unroll
        for (int nt = 0; nt < 2; ++nt) {
            const int nb = waveN * 32 + nt * 16 + bRow;
            U16x16 th, tl;
#pragma unroll
            for (int e = 0; e < 16; ++e) {
                const int kk = 8 * khalf + e + (e >= 8 ? 8 : 0);
                th.v[e] = BsHi[cur][kk * 72 + nb];
                tl.v[e] = BsLo[cur][kk * 72 + nb];
            }
            bHi[nt] = __builtin_bit_cast(v16bf, th);
            bLo[nt] = __builtin_bit_cast(v16bf, tl);
        }
        // ---- bf16x3 accumulation: hi*hi + hi*lo + lo*hi
#pragma unroll
        for (int mt = 0; mt < 2; ++mt)
#pragma unroll
            for (int nt = 0; nt < 2; ++nt) {
                acc[mt][nt] = __builtin_amdgcn_wmma_f32_16x16x32_bf16(
                    false, aHi[mt], false, bHi[nt], (short)0, acc[mt][nt], false, false);
                acc[mt][nt] = __builtin_amdgcn_wmma_f32_16x16x32_bf16(
                    false, aHi[mt], false, bLo[nt], (short)0, acc[mt][nt], false, false);
                acc[mt][nt] = __builtin_amdgcn_wmma_f32_16x16x32_bf16(
                    false, aLo[mt], false, bHi[nt], (short)0, acc[mt][nt], false, false);
            }
        cur ^= 1;
    }

    // ---- epilogue (C 16x16 f32: M = r + 8*(lane>=16), N = lane&15)
    const int cCol = lane & 15;
    const int cRow = 8 * (lane >> 4);
#pragma unroll
    for (int mt = 0; mt < 2; ++mt)
#pragma unroll
        for (int nt = 0; nt < 2; ++nt)
#pragma unroll
            for (int r = 0; r < 8; ++r) {
                const int gm = Mbase + waveM * 32 + mt * 16 + cRow + r;
                const int gn = Nbase + waveN * 32 + nt * 16 + cCol;
                const float v = acc[mt][nt][r] * scale;
                if (OUT_SPLIT) {
                    const unsigned short h = f32_to_bf16(v);
                    outHi[(size_t)gm * ldo + gn] = h;
                    outLo[(size_t)gm * ldo + gn] = f32_to_bf16(v - bf16_to_f32(h));
                } else {
                    outF[(size_t)gm * ldo + gn] = v;
                }
            }
}

// ---------------------------------------------------------------------------
// Row softmax over 4096 columns, one 256-thread block per row
// ---------------------------------------------------------------------------
__global__ __launch_bounds__(256)
void softmax_rows_kernel(float* __restrict__ data) {
    __shared__ float red[256];
    const int tid = threadIdx.x;
    float* rowp = data + (size_t)blockIdx.x * SEQ_LEN;

    float vals[16];
    float m = -3.402823466e38f;
#pragma unroll
    for (int i = 0; i < 16; ++i) {
        vals[i] = rowp[i * 256 + tid];
        m = fmaxf(m, vals[i]);
    }
    red[tid] = m;
    __syncthreads();
    for (int s = 128; s > 0; s >>= 1) {
        if (tid < s) red[tid] = fmaxf(red[tid], red[tid + s]);
        __syncthreads();
    }
    m = red[0];
    __syncthreads();

    float sum = 0.0f;
#pragma unroll
    for (int i = 0; i < 16; ++i) {
        vals[i] = __expf(vals[i] - m);
        sum += vals[i];
    }
    red[tid] = sum;
    __syncthreads();
    for (int s = 128; s > 0; s >>= 1) {
        if (tid < s) red[tid] += red[tid + s];
        __syncthreads();
    }
    const float inv = 1.0f / red[0];
#pragma unroll
    for (int i = 0; i < 16; ++i) rowp[i * 256 + tid] = vals[i] * inv;
}

// ---------------------------------------------------------------------------
extern "C" void kernel_launch(void* const* d_in, const int* in_sizes, int n_in,
                              void* d_out, int out_size, void* d_ws, size_t ws_size,
                              hipStream_t stream) {
    (void)in_sizes; (void)n_in; (void)out_size; (void)ws_size;
    const float* x   = (const float*)d_in[0];
    const float* w_q = (const float*)d_in[1];
    const float* w_k = (const float*)d_in[2];
    // d_in[3] (w_v), d_in[4] (out_proj) unused by the reference output.
    float* out = (float*)d_out;

    // workspace carve-up (~58.7 MB total)
    char* ws = (char*)d_ws;
    size_t off = 0;
    auto carve = [&](size_t bytes) { char* p = ws + off; off += bytes; return p; };
    const size_t xN = (size_t)SEQ_LEN * DMODEL;   // 4194304
    const size_t wN = (size_t)DMODEL * DMODEL;    // 1048576
    unsigned short* x_hi  = (unsigned short*)carve(xN * 2);
    unsigned short* x_lo  = (unsigned short*)carve(xN * 2);
    unsigned short* wq_hi = (unsigned short*)carve(wN * 2);
    unsigned short* wq_lo = (unsigned short*)carve(wN * 2);
    unsigned short* wk_hi = (unsigned short*)carve(wN * 2);
    unsigned short* wk_lo = (unsigned short*)carve(wN * 2);
    unsigned short* q_hi  = (unsigned short*)carve(xN * 2);
    unsigned short* q_lo  = (unsigned short*)carve(xN * 2);
    unsigned short* k_hi  = (unsigned short*)carve(xN * 2);
    unsigned short* k_lo  = (unsigned short*)carve(xN * 2);

    // 0) split inputs into bf16 hi/lo
    split_bf16_kernel<<<4096, 256, 0, stream>>>(x,   x_hi, x_lo, (int)xN);
    split_bf16_kernel<<<1024, 256, 0, stream>>>(w_q, wq_hi, wq_lo, (int)wN);
    split_bf16_kernel<<<1024, 256, 0, stream>>>(w_k, wk_hi, wk_lo, (int)wN);

    // 1) Q = x @ w_q, K = x @ w_k   (output as bf16 hi/lo split)
    dim3 gridProj(SEQ_LEN / 128, DMODEL / 64);
    gemm_bf16x3_kernel<false, true><<<gridProj, 256, 0, stream>>>(
        x_hi, x_lo, wq_hi, wq_lo, DMODEL, DMODEL, DMODEL, DMODEL, 1.0f,
        nullptr, q_hi, q_lo);
    gemm_bf16x3_kernel<false, true><<<gridProj, 256, 0, stream>>>(
        x_hi, x_lo, wk_hi, wk_lo, DMODEL, DMODEL, DMODEL, DMODEL, 1.0f,
        nullptr, k_hi, k_lo);

    // 2) scores = (Q @ K^T) / 32  -> d_out f32
    dim3 gridScores(SEQ_LEN / 128, SEQ_LEN / 64);
    gemm_bf16x3_kernel<true, false><<<gridScores, 256, 0, stream>>>(
        q_hi, q_lo, k_hi, k_lo, DMODEL, DMODEL, DMODEL, SEQ_LEN, 0.03125f,
        out, nullptr, nullptr);

    // 3) row softmax in place
    softmax_rows_kernel<<<SEQ_LEN, 256, 0, stream>>>(out);
}
